// GATEncoder_57973468562008
// MI455X (gfx1250) — matrix operand order repaired and checked by reading, not compiled, not run on hardware
//
#include <hip/hip_runtime.h>

// ---------------------------------------------------------------------------
// GAT encoder for MI455X (gfx1250, wave32, WMMA, TDM).
//   1. convert x -> bf16; transpose-convert W1/W2/W3 -> bf16 [N][K]
//   2. WMMA GEMM: Wh1 = x @ W1 (bf16 in, f32 acc, bf16 out)
//   3. f_src/f_dst = Wh @ a  (wave-per-row dot)            + global max(f_dst)
//   4. masked-softmax attention h = relu(P1 @ Wh1): adjacency tiles streamed
//      by the Tensor Data Mover into padded LDS (double buffered, TENSORcnt),
//      P built directly in WMMA A-layout registers, Wh staged transposed.
//   5. WMMA GEMMs: Wh2 = h @ W2, Wh3 = h @ W3
//   6. fused layer-2/3 attention (single adj pass): mu, exp(log_sig)
// Stabilizer m_i = leakyrelu(fs_i + max_j fd_j) is the exact unmasked row max
// (monotone), so no online rescaling; KSPLIT partials combine by addition
// (deterministic, no float atomics).
// ---------------------------------------------------------------------------

#define NN 8192
#define NFEAT 512
#define NHID 256
#define NLAT 64
#define KSPLIT 4
#define GAT_ALPHA 0.2f
#define GAT_NEG_INF -9.0e15f

#if defined(__HIP_DEVICE_COMPILE__) && __has_builtin(__builtin_amdgcn_tensor_load_to_lds)
#define USE_TDM 1
#else
#define USE_TDM 0
#endif

typedef __attribute__((ext_vector_type(16))) __bf16 bf16x16;
typedef __attribute__((ext_vector_type(8)))  float  f32x8;

union AB16 {
    bf16x16 v;
    unsigned short u[16];
    uint4 q[2];
};

__device__ __forceinline__ unsigned short f2bf(float f) {
    union { float f; unsigned u; } c; c.f = f;
    unsigned r = c.u + 0x7FFFu + ((c.u >> 16) & 1u);   // RNE
    return (unsigned short)(r >> 16);
}
__device__ __forceinline__ float bf2f(unsigned short s) {
    union { unsigned u; float f; } c; c.u = ((unsigned)s) << 16;
    return c.f;
}
__device__ __forceinline__ f32x8 wmma_bf16(const AB16& a, const AB16& b, f32x8 c) {
    return __builtin_amdgcn_wmma_f32_16x16x32_bf16(false, a.v, false, b.v,
                                                   (short)0, c, false, false);
}
__device__ __forceinline__ float lrelu(float x) { return x > 0.f ? x : GAT_ALPHA * x; }

#if USE_TDM
typedef __attribute__((ext_vector_type(4))) unsigned int u32x4;
typedef __attribute__((ext_vector_type(8))) int i32x8;
typedef __attribute__((ext_vector_type(4))) int i32x4;

// TDM: DMA a [rows x 32] f32 tile (row stride NN) into LDS with 4-DWORD pad
// every 32 DWORDs -> LDS rows of 36 floats (bank-staggered). D# per ISA 8.3/8.4.
// 6-arg builtin form (clang-23 / therock): (g0, g1, g2, g3, g4, cpol).
__device__ __forceinline__ void tdm_load_f32_tile32(const float* gptr, unsigned lds_off,
                                                    int rows) {
    unsigned long long ga = (unsigned long long)(const void*)gptr;
    u32x4 g0;
    g0[0] = 1u;                                   // count=1 (valid user descriptor)
    g0[1] = lds_off;                              // lds_addr (bytes)
    g0[2] = (unsigned)(ga & 0xFFFFFFFFu);         // global_addr[31:0]
    g0[3] = (unsigned)((ga >> 32) & 0x1FFFFFFu)   // global_addr[56:32]
          | (2u << 30);                           // type = 2 ("image")
    i32x8 g1;
    g1[0] = (2 << 16)                             // data_size = 4 bytes
          | (1 << 20)                             // pad_enable
          | (4 << 22)                             // pad_interval: 32 DWORDs
          | (3 << 25);                            // pad_amount: 4 DWORDs
    g1[1] = (NN & 0xFFFF) << 16;                  // tensor_dim0[15:0] @ bits 63:48
    g1[2] = ((NN >> 16) & 0xFFFF)                 // tensor_dim0[31:16] @ bits 79:64
          | ((NN & 0xFFFF) << 16);                // tensor_dim1[15:0] @ bits 95:80
    g1[3] = ((NN >> 16) & 0xFFFF)                 // tensor_dim1[31:16] @ bits 111:96
          | (32 << 16);                           // tile_dim0 = 32 @ bits 127:112
    g1[4] = rows & 0xFFFF;                        // tile_dim1; tile_dim2 = 0
    g1[5] = NN;                                   // tensor_dim0_stride[31:0]
    g1[6] = 0;                                    // stride hi, dim1_stride lo
    g1[7] = 0;
    const i32x4 z4 = {0, 0, 0, 0};
    const i32x8 z8 = {0, 0, 0, 0, 0, 0, 0, 0};
    __builtin_amdgcn_tensor_load_to_lds(g0, g1, z4, z4, z8, 0);
}
#endif

// ---------------------------------------------------------------------------
__global__ void k_cvt_bf16(const float* __restrict__ in, unsigned short* __restrict__ out, int n) {
    for (int i = blockIdx.x * blockDim.x + threadIdx.x; i < n; i += gridDim.x * blockDim.x)
        out[i] = f2bf(in[i]);
}

// transpose + convert: in[R][C] f32 -> out[C][R] bf16
__global__ void k_transpose_bf16(const float* __restrict__ in, unsigned short* __restrict__ out,
                                 int R, int C) {
    int n = R * C;
    for (int i = blockIdx.x * blockDim.x + threadIdx.x; i < n; i += gridDim.x * blockDim.x) {
        int r = i / C, c = i - r * C;
        out[(size_t)c * R + r] = f2bf(in[i]);
    }
}

// ---------------------------------------------------------------------------
// C[M][Nf](bf16) = A[M][K](bf16) @ BT[Nf][K](bf16), f32 accumulate.
__global__ void k_gemm_bf16(const unsigned short* __restrict__ A,
                            const unsigned short* __restrict__ BT,
                            unsigned short* __restrict__ C,
                            int M, int K, int Nf) {
    const int lane = threadIdx.x & 31;
    const int wave = threadIdx.x >> 5;
    const int m16  = lane & 15;
    const int half = lane >> 4;
    const int nt64 = Nf >> 6;
    const int wid  = blockIdx.x * (blockDim.x >> 5) + wave;
    const int row0 = (wid / nt64) * 16;
    const int n0   = (wid % nt64) * 64;
    if (row0 >= M) return;  // wave-uniform

    f32x8 acc[4];
    const f32x8 zro = {0.f, 0.f, 0.f, 0.f, 0.f, 0.f, 0.f, 0.f};
#pragma unroll
    for (int t = 0; t < 4; ++t) acc[t] = zro;

    for (int kb = 0; kb < (K >> 5); ++kb) {
        AB16 a;
        const unsigned short* ap = A + (size_t)(row0 + m16) * K + kb * 32 + half * 8;
        a.q[0] = *(const uint4*)ap;
        a.q[1] = *(const uint4*)(ap + 16);
#pragma unroll
        for (int t = 0; t < 4; ++t) {
            AB16 b;
            const unsigned short* bp = BT + (size_t)(n0 + t * 16 + m16) * K + kb * 32 + half * 8;
            b.q[0] = *(const uint4*)bp;
            b.q[1] = *(const uint4*)(bp + 16);
            acc[t] = wmma_bf16(a, b, acc[t]);
        }
    }
#pragma unroll
    for (int t = 0; t < 4; ++t)
#pragma unroll
        for (int j = 0; j < 8; ++j)
            C[(size_t)(row0 + half * 8 + j) * Nf + n0 + t * 16 + m16] = f2bf(acc[t][j]);
}

// ---------------------------------------------------------------------------
template <int F>
__global__ void k_fvec(const unsigned short* __restrict__ Wh, const float* __restrict__ a,
                       float* __restrict__ fs, float* __restrict__ fd) {
    const int lane = threadIdx.x & 31;
    const int r = blockIdx.x * (blockDim.x >> 5) + (threadIdx.x >> 5);
    if (r >= NN) return;
    float ss = 0.f, sd = 0.f;
    for (int i = lane; i < F; i += 32) {
        float w = bf2f(Wh[(size_t)r * F + i]);
        ss += w * a[i];
        sd += w * a[F + i];
    }
#pragma unroll
    for (int o = 16; o; o >>= 1) {
        ss += __shfl_xor(ss, o, 32);
        sd += __shfl_xor(sd, o, 32);
    }
    if (lane == 0) { fs[r] = ss; fd[r] = sd; }
}

__global__ void k_maxreduce(const float* __restrict__ v, int n, float* __restrict__ out) {
    __shared__ float s[256];
    float m = -3.4e38f;
    for (int i = threadIdx.x; i < n; i += 256) m = fmaxf(m, v[i]);
    s[threadIdx.x] = m;
    __syncthreads();
    for (int o = 128; o; o >>= 1) {
        if (threadIdx.x < o) s[threadIdx.x] = fmaxf(s[threadIdx.x], s[threadIdx.x + o]);
        __syncthreads();
    }
    if (threadIdx.x == 0) out[0] = s[0];
}

// ---------------------------------------------------------------------------
#define LSTR 40    // Wh LDS halfword stride: 80B rows, 16B-aligned, bank-staggered
#define ASTR 36    // adj LDS float stride (TDM pad: 32 DW + 4 DW)

// Layer-1 attention: pnum[split][r][F] = sum_k P(16x32) @ Wh(32xF), pden = row sums.
template <int F>
__global__ void k_attn1(const float* __restrict__ adj,
                        const float* __restrict__ fs, const float* __restrict__ fd,
                        const float* __restrict__ fdmax,
                        const unsigned short* __restrict__ Whbf,
                        float* __restrict__ pnum, float* __restrict__ pden) {
    __shared__ unsigned short whT[F * LSTR];
#if USE_TDM
    __shared__ float adjT[2][64 * ASTR];
#endif
    const int tid  = threadIdx.x;
    const int lane = tid & 31;
    const int wave = tid >> 5;
    const int m16  = lane & 15;
    const int half = lane >> 4;
    const int rtb  = blockIdx.x * 64;          // 64 rows per workgroup
    const int row0 = rtb + wave * 16;
    const int r    = row0 + m16;
    const int split = blockIdx.y;
    const int colsPer = NN / KSPLIT;
    const int cbase = split * colsPer;
    const int nb = colsPer / 32;

    const float fsm = fs[r];
    const float mi  = lrelu(fsm + fdmax[0]);   // exact unmasked row max

    f32x8 acc[F / 16];
    const f32x8 zro = {0.f, 0.f, 0.f, 0.f, 0.f, 0.f, 0.f, 0.f};
#pragma unroll
    for (int t = 0; t < F / 16; ++t) acc[t] = zro;
    float rsum = 0.f;

#if USE_TDM
    if (wave == 0)   // prologue: DMA first adjacency tile
        tdm_load_f32_tile32(adj + (size_t)rtb * NN + cbase,
                            (unsigned)(unsigned long long)(const void*)&adjT[0][0], 64);
#endif

    for (int kb = 0; kb < nb; ++kb) {
        const int cb = cbase + kb * 32;
        __syncthreads();   // previous block's LDS consumers done
#if USE_TDM
        if (wave == 0) {
            if (kb + 1 < nb) {
                tdm_load_f32_tile32(adj + (size_t)rtb * NN + cb + 32,
                                    (unsigned)(unsigned long long)(const void*)&adjT[(kb + 1) & 1][0],
                                    64);
                __builtin_amdgcn_s_wait_tensorcnt(1);   // tile kb landed
            } else {
                __builtin_amdgcn_s_wait_tensorcnt(0);
            }
        }
#endif
        // stage Wh[cb..cb+31][0..F) transposed into LDS; prefetch next block
        for (int g = tid; g < (32 * F) / 8; g += blockDim.x) {
            int kk = g / (F / 8);
            int fb = (g % (F / 8)) * 8;
            uint4 qv = *(const uint4*)(Whbf + (size_t)(cb + kk) * F + fb);
            if (kb + 1 < nb)
                __builtin_prefetch(Whbf + (size_t)(cb + 32 + kk) * F + fb, 0, 1);
            const unsigned short* u = (const unsigned short*)&qv;
#pragma unroll
            for (int i = 0; i < 8; ++i) whT[(fb + i) * LSTR + kk] = u[i];
        }
        __syncthreads();   // whT staged + adjacency tile kb ready

        // build P tile directly in WMMA A-layout (lane row m16, K-halves by half)
        AB16 pa;
#if USE_TDM
        const float* arow = &adjT[kb & 1][(wave * 16 + m16) * ASTR + half * 8];
#else
        const float* arow = adj + (size_t)r * NN + cb + half * 8;
#endif
        const float* fdp = fd + cb + half * 8;
#pragma unroll
        for (int c = 0; c < 2; ++c) {
            const int off = c * 16;
            float4 av0 = *(const float4*)(arow + off);
            float4 av1 = *(const float4*)(arow + off + 4);
            float4 dv0 = *(const float4*)(fdp + off);
            float4 dv1 = *(const float4*)(fdp + off + 4);
            float aa[8] = {av0.x, av0.y, av0.z, av0.w, av1.x, av1.y, av1.z, av1.w};
            float dd[8] = {dv0.x, dv0.y, dv0.z, dv0.w, dv1.x, dv1.y, dv1.z, dv1.w};
#pragma unroll
            for (int e = 0; e < 8; ++e) {
                float xv = lrelu(fsm + dd[e]);
                xv = (aa[e] > 0.f) ? xv : GAT_NEG_INF;
                float p = __expf(xv - mi);
                rsum += p;
                pa.u[c * 8 + e] = f2bf(p);
            }
        }
#pragma unroll
        for (int t = 0; t < F / 16; ++t) {
            AB16 b;
            const unsigned short* lp = whT + (t * 16 + m16) * LSTR + half * 8;
            b.q[0] = *(const uint4*)lp;
            b.q[1] = *(const uint4*)(lp + 16);
            acc[t] = wmma_bf16(pa, b, acc[t]);
        }
    }

    rsum += __shfl_xor(rsum, 16, 32);
    if (half == 0) pden[(size_t)split * NN + r] = rsum;
    const int orow = row0 + half * 8;
#pragma unroll
    for (int t = 0; t < F / 16; ++t)
#pragma unroll
        for (int j = 0; j < 8; ++j)
            pnum[((size_t)split * NN + orow + j) * F + t * 16 + m16] = acc[t][j];
}

// ---------------------------------------------------------------------------
// Fused layer-2/3 attention: one adjacency pass feeds both heads (F=64 each).
__global__ void k_attn23(const float* __restrict__ adj,
                         const float* __restrict__ fs2, const float* __restrict__ fd2,
                         const float* __restrict__ fx2, const unsigned short* __restrict__ Wh2,
                         const float* __restrict__ fs3, const float* __restrict__ fd3,
                         const float* __restrict__ fx3, const unsigned short* __restrict__ Wh3,
                         float* __restrict__ pn2, float* __restrict__ pd2,
                         float* __restrict__ pn3, float* __restrict__ pd3) {
    constexpr int F = 64;
    __shared__ unsigned short wh2T[F * LSTR];
    __shared__ unsigned short wh3T[F * LSTR];
#if USE_TDM
    __shared__ float adjT[2][64 * ASTR];
#endif
    const int tid  = threadIdx.x;
    const int lane = tid & 31;
    const int wave = tid >> 5;
    const int m16  = lane & 15;
    const int half = lane >> 4;
    const int rtb  = blockIdx.x * 64;
    const int row0 = rtb + wave * 16;
    const int r    = row0 + m16;
    const int split = blockIdx.y;
    const int colsPer = NN / KSPLIT;
    const int cbase = split * colsPer;
    const int nb = colsPer / 32;

    const float fsm2 = fs2[r], fsm3 = fs3[r];
    const float mi2 = lrelu(fsm2 + fx2[0]);
    const float mi3 = lrelu(fsm3 + fx3[0]);

    f32x8 acc2[4], acc3[4];
    const f32x8 zro = {0.f, 0.f, 0.f, 0.f, 0.f, 0.f, 0.f, 0.f};
#pragma unroll
    for (int t = 0; t < 4; ++t) { acc2[t] = zro; acc3[t] = zro; }
    float rs2 = 0.f, rs3 = 0.f;

#if USE_TDM
    if (wave == 0)
        tdm_load_f32_tile32(adj + (size_t)rtb * NN + cbase,
                            (unsigned)(unsigned long long)(const void*)&adjT[0][0], 64);
#endif

    for (int kb = 0; kb < nb; ++kb) {
        const int cb = cbase + kb * 32;
        __syncthreads();
#if USE_TDM
        if (wave == 0) {
            if (kb + 1 < nb) {
                tdm_load_f32_tile32(adj + (size_t)rtb * NN + cb + 32,
                                    (unsigned)(unsigned long long)(const void*)&adjT[(kb + 1) & 1][0],
                                    64);
                __builtin_amdgcn_s_wait_tensorcnt(1);
            } else {
                __builtin_amdgcn_s_wait_tensorcnt(0);
            }
        }
#endif
        for (int g = tid; g < (32 * F) / 8; g += blockDim.x) {
            int kk = g / (F / 8);
            int fb = (g % (F / 8)) * 8;
            uint4 q2 = *(const uint4*)(Wh2 + (size_t)(cb + kk) * F + fb);
            uint4 q3 = *(const uint4*)(Wh3 + (size_t)(cb + kk) * F + fb);
            if (kb + 1 < nb) {
                __builtin_prefetch(Wh2 + (size_t)(cb + 32 + kk) * F + fb, 0, 1);
                __builtin_prefetch(Wh3 + (size_t)(cb + 32 + kk) * F + fb, 0, 1);
            }
            const unsigned short* u2 = (const unsigned short*)&q2;
            const unsigned short* u3 = (const unsigned short*)&q3;
#pragma unroll
            for (int i = 0; i < 8; ++i) {
                wh2T[(fb + i) * LSTR + kk] = u2[i];
                wh3T[(fb + i) * LSTR + kk] = u3[i];
            }
        }
        __syncthreads();

        AB16 pa2, pa3;
#if USE_TDM
        const float* arow = &adjT[kb & 1][(wave * 16 + m16) * ASTR + half * 8];
#else
        const float* arow = adj + (size_t)r * NN + cb + half * 8;
#endif
        const float* d2p = fd2 + cb + half * 8;
        const float* d3p = fd3 + cb + half * 8;
#pragma unroll
        for (int c = 0; c < 2; ++c) {
            const int off = c * 16;
            float4 av0 = *(const float4*)(arow + off);
            float4 av1 = *(const float4*)(arow + off + 4);
            float4 e20 = *(const float4*)(d2p + off);
            float4 e21 = *(const float4*)(d2p + off + 4);
            float4 e30 = *(const float4*)(d3p + off);
            float4 e31 = *(const float4*)(d3p + off + 4);
            float aa[8]  = {av0.x, av0.y, av0.z, av0.w, av1.x, av1.y, av1.z, av1.w};
            float d2v[8] = {e20.x, e20.y, e20.z, e20.w, e21.x, e21.y, e21.z, e21.w};
            float d3v[8] = {e30.x, e30.y, e30.z, e30.w, e31.x, e31.y, e31.z, e31.w};
#pragma unroll
            for (int e = 0; e < 8; ++e) {
                bool conn = aa[e] > 0.f;
                float x2 = lrelu(fsm2 + d2v[e]); x2 = conn ? x2 : GAT_NEG_INF;
                float x3 = lrelu(fsm3 + d3v[e]); x3 = conn ? x3 : GAT_NEG_INF;
                float p2 = __expf(x2 - mi2);
                float p3 = __expf(x3 - mi3);
                rs2 += p2; rs3 += p3;
                pa2.u[c * 8 + e] = f2bf(p2);
                pa3.u[c * 8 + e] = f2bf(p3);
            }
        }
#pragma unroll
        for (int t = 0; t < 4; ++t) {
            AB16 b;
            const unsigned short* lp = wh2T + (t * 16 + m16) * LSTR + half * 8;
            b.q[0] = *(const uint4*)lp;
            b.q[1] = *(const uint4*)(lp + 16);
            acc2[t] = wmma_bf16(pa2, b, acc2[t]);
        }
#pragma unroll
        for (int t = 0; t < 4; ++t) {
            AB16 b;
            const unsigned short* lp = wh3T + (t * 16 + m16) * LSTR + half * 8;
            b.q[0] = *(const uint4*)lp;
            b.q[1] = *(const uint4*)(lp + 16);
            acc3[t] = wmma_bf16(pa3, b, acc3[t]);
        }
    }

    rs2 += __shfl_xor(rs2, 16, 32);
    rs3 += __shfl_xor(rs3, 16, 32);
    if (half == 0) {
        pd2[(size_t)split * NN + r] = rs2;
        pd3[(size_t)split * NN + r] = rs3;
    }
    const int orow = row0 + half * 8;
#pragma unroll
    for (int t = 0; t < 4; ++t)
#pragma unroll
        for (int j = 0; j < 8; ++j) {
            pn2[((size_t)split * NN + orow + j) * F + t * 16 + m16] = acc2[t][j];
            pn3[((size_t)split * NN + orow + j) * F + t * 16 + m16] = acc3[t][j];
        }
}

// ---------------------------------------------------------------------------
__global__ void k_fin1(const float* __restrict__ pnum, const float* __restrict__ pden,
                       unsigned short* __restrict__ hbf) {
    const int n = NN * NHID;
    for (int i = blockIdx.x * blockDim.x + threadIdx.x; i < n; i += gridDim.x * blockDim.x) {
        int r = i >> 8;
        float nu = 0.f, de = 0.f;
#pragma unroll
        for (int s = 0; s < KSPLIT; ++s) {
            nu += pnum[(size_t)s * NN * NHID + i];
            de += pden[(size_t)s * NN + r];
        }
        float h = nu / de;
        hbf[i] = f2bf(h > 0.f ? h : 0.f);
    }
}

__global__ void k_fin23(const float* __restrict__ pn2, const float* __restrict__ pd2,
                        const float* __restrict__ pn3, const float* __restrict__ pd3,
                        float* __restrict__ out) {
    const int n = NN * NLAT;
    for (int i = blockIdx.x * blockDim.x + threadIdx.x; i < n; i += gridDim.x * blockDim.x) {
        int r = i >> 6;
        float n2 = 0.f, d2 = 0.f, n3 = 0.f, d3 = 0.f;
#pragma unroll
        for (int s = 0; s < KSPLIT; ++s) {
            n2 += pn2[(size_t)s * NN * NLAT + i];
            d2 += pd2[(size_t)s * NN + r];
            n3 += pn3[(size_t)s * NN * NLAT + i];
            d3 += pd3[(size_t)s * NN + r];
        }
        out[i] = n2 / d2;                            // mu
        out[(size_t)NN * NLAT + i] = expf(n3 / d3);  // exp(log_sigma)
    }
}

// ---------------------------------------------------------------------------
static constexpr size_t OFF_XBF = 0;
static constexpr size_t OFF_W1T = OFF_XBF + (size_t)NN * NFEAT * 2;
static constexpr size_t OFF_W2T = OFF_W1T + (size_t)NFEAT * NHID * 2;
static constexpr size_t OFF_W3T = OFF_W2T + (size_t)NHID * NLAT * 2;
static constexpr size_t OFF_WH1 = OFF_W3T + (size_t)NHID * NLAT * 2;
static constexpr size_t OFF_HBF = OFF_WH1 + (size_t)NN * NHID * 2;
static constexpr size_t OFF_WH2 = OFF_HBF + (size_t)NN * NHID * 2;
static constexpr size_t OFF_WH3 = OFF_WH2 + (size_t)NN * NLAT * 2;
static constexpr size_t OFF_FS1 = OFF_WH3 + (size_t)NN * NLAT * 2;
static constexpr size_t OFF_FD1 = OFF_FS1 + (size_t)NN * 4;
static constexpr size_t OFF_FS2 = OFF_FD1 + (size_t)NN * 4;
static constexpr size_t OFF_FD2 = OFF_FS2 + (size_t)NN * 4;
static constexpr size_t OFF_FS3 = OFF_FD2 + (size_t)NN * 4;
static constexpr size_t OFF_FD3 = OFF_FS3 + (size_t)NN * 4;
static constexpr size_t OFF_SC  = OFF_FD3 + (size_t)NN * 4;
static constexpr size_t OFF_PD1 = OFF_SC + 256;
static constexpr size_t OFF_PD2 = OFF_PD1 + (size_t)KSPLIT * NN * 4;
static constexpr size_t OFF_PD3 = OFF_PD2 + (size_t)KSPLIT * NN * 4;
static constexpr size_t OFF_PN1 = OFF_PD3 + (size_t)KSPLIT * NN * 4;
static constexpr size_t OFF_PN2 = OFF_PN1;  // reused after layer-1 finalize
static constexpr size_t OFF_PN3 = OFF_PN2 + (size_t)KSPLIT * NN * NLAT * 4;

extern "C" void kernel_launch(void* const* d_in, const int* in_sizes, int n_in,
                              void* d_out, int out_size, void* d_ws, size_t ws_size,
                              hipStream_t stream) {
    const float* x   = (const float*)d_in[0];
    const float* adj = (const float*)d_in[1];
    const float* W1  = (const float*)d_in[2];
    const float* a1  = (const float*)d_in[3];
    const float* W2  = (const float*)d_in[4];
    const float* a2  = (const float*)d_in[5];
    const float* W3  = (const float*)d_in[6];
    const float* a3  = (const float*)d_in[7];
    float* out = (float*)d_out;
    char*  w   = (char*)d_ws;

    unsigned short* XBF = (unsigned short*)(w + OFF_XBF);
    unsigned short* W1T = (unsigned short*)(w + OFF_W1T);
    unsigned short* W2T = (unsigned short*)(w + OFF_W2T);
    unsigned short* W3T = (unsigned short*)(w + OFF_W3T);
    unsigned short* WH1 = (unsigned short*)(w + OFF_WH1);
    unsigned short* HBF = (unsigned short*)(w + OFF_HBF);
    unsigned short* WH2 = (unsigned short*)(w + OFF_WH2);
    unsigned short* WH3 = (unsigned short*)(w + OFF_WH3);
    float* FS1 = (float*)(w + OFF_FS1);
    float* FD1 = (float*)(w + OFF_FD1);
    float* FS2 = (float*)(w + OFF_FS2);
    float* FD2 = (float*)(w + OFF_FD2);
    float* FS3 = (float*)(w + OFF_FS3);
    float* FD3 = (float*)(w + OFF_FD3);
    float* SC  = (float*)(w + OFF_SC);
    float* PD1 = (float*)(w + OFF_PD1);
    float* PD2 = (float*)(w + OFF_PD2);
    float* PD3 = (float*)(w + OFF_PD3);
    float* PN1 = (float*)(w + OFF_PN1);
    float* PN2 = (float*)(w + OFF_PN2);
    float* PN3 = (float*)(w + OFF_PN3);

    k_cvt_bf16<<<2048, 256, 0, stream>>>(x, XBF, NN * NFEAT);
    k_transpose_bf16<<<512, 256, 0, stream>>>(W1, W1T, NFEAT, NHID);
    k_transpose_bf16<<<64, 256, 0, stream>>>(W2, W2T, NHID, NLAT);
    k_transpose_bf16<<<64, 256, 0, stream>>>(W3, W3T, NHID, NLAT);

    k_gemm_bf16<<<512, 128, 0, stream>>>(XBF, W1T, WH1, NN, NFEAT, NHID);

    k_fvec<NHID><<<NN / 4, 128, 0, stream>>>(WH1, a1, FS1, FD1);
    k_maxreduce<<<1, 256, 0, stream>>>(FD1, NN, SC + 0);

    {
        dim3 g(NN / 64, KSPLIT);
        k_attn1<NHID><<<g, 128, 0, stream>>>(adj, FS1, FD1, SC + 0, WH1, PN1, PD1);
    }
    k_fin1<<<2048, 256, 0, stream>>>(PN1, PD1, HBF);

    k_gemm_bf16<<<128, 128, 0, stream>>>(HBF, W2T, WH2, NN, NHID, NLAT);
    k_gemm_bf16<<<128, 128, 0, stream>>>(HBF, W3T, WH3, NN, NHID, NLAT);

    k_fvec<NLAT><<<NN / 4, 128, 0, stream>>>(WH2, a2, FS2, FD2);
    k_fvec<NLAT><<<NN / 4, 128, 0, stream>>>(WH3, a3, FS3, FD3);
    k_maxreduce<<<1, 256, 0, stream>>>(FD2, NN, SC + 1);
    k_maxreduce<<<1, 256, 0, stream>>>(FD3, NN, SC + 2);

    {
        dim3 g(NN / 64, KSPLIT);
        k_attn23<<<g, 128, 0, stream>>>(adj, FS2, FD2, SC + 1, WH2,
                                        FS3, FD3, SC + 2, WH3,
                                        PN2, PD2, PN3, PD3);
    }
    k_fin23<<<512, 256, 0, stream>>>(PN2, PD2, PN3, PD3, out);
}
